// GCN_83837761618292
// MI455X (gfx1250) — matrix operand order repaired and checked by reading, not compiled, not run on hardware
//
#include <hip/hip_runtime.h>

// Problem constants (match reference)
#define NN   8192      // nodes
#define NE   16384     // edges (= conv2 output width)
#define DF   64        // feature / hidden dim
#define CC   320       // combined dim = 64*(1+4)

typedef __attribute__((ext_vector_type(16))) _Float16   v16h;
typedef __attribute__((ext_vector_type(8)))  float      v8f;
typedef __attribute__((ext_vector_type(8)))  unsigned int v8u;

union HU { v8u u; v16h h; };

__device__ inline unsigned int pack2h(float a, float b) {
  union { _Float16 h; unsigned short s; } ua, ub;
  ua.h = (_Float16)a; ub.h = (_Float16)b;
  return (unsigned int)ua.s | ((unsigned int)ub.s << 16);
}

// ---- degree / normalization -------------------------------------------------
__global__ void k_deg_init(float* __restrict__ deg) {
  int i = blockIdx.x * 256 + threadIdx.x;
  if (i < NN) deg[i] = 1.0f;                      // self-loop weight pre-added
}

__global__ void k_deg_scatter(const int* __restrict__ ei,
                              const float* __restrict__ ew,
                              float* __restrict__ deg) {
  int e = blockIdx.x * 256 + threadIdx.x;
  if (e < NE) atomicAdd(&deg[ei[NE + e]], ew[e]); // col = ei[E+e]
}

__global__ void k_dis(float* __restrict__ deg) {
  int i = blockIdx.x * 256 + threadIdx.x;
  if (i < NN) deg[i] = rsqrtf(deg[i]);            // deg >= 1 always
}

// ---- conv1 GEMM: x1 = gathered(combined) @ W1  ([N,320]@[320,64]) ----------
// 256 threads/block, 32 nodes/block; W1 streamed through LDS one 64x64 slice
// per gather-source s (s=0: self, s=1..4: neighbours).
__global__ void k_x1(const float* __restrict__ emb, const int* __restrict__ nodes,
                     const int* __restrict__ nbr, const float* __restrict__ W1,
                     float* __restrict__ x1) {
  __shared__ float sW[64 * 64];                   // 16 KB
  const int tid = threadIdx.x;
  const int h   = tid & 63;
  const int sub = tid >> 6;                       // 0..3
  const int base = blockIdx.x * 32;
  float acc[8];
#pragma unroll
  for (int p = 0; p < 8; ++p) acc[p] = 0.f;

  for (int s = 0; s < 5; ++s) {
    __syncthreads();
    for (int idx = tid; idx < 4096; idx += 256) sW[idx] = W1[s * 4096 + idx];
    __syncthreads();
#pragma unroll
    for (int p = 0; p < 8; ++p) {
      int i   = base + p * 4 + sub;
      int src = (s == 0) ? nodes[i] : nodes[nbr[i * 4 + (s - 1)]];
      const float* er = emb + (size_t)src * DF;
      float a = 0.f;
#pragma unroll
      for (int d = 0; d < DF; ++d) a += er[d] * sW[d * 64 + h];
      acc[p] += a;
    }
  }
#pragma unroll
  for (int p = 0; p < 8; ++p) {
    int i = base + p * 4 + sub;
    x1[(size_t)i * DF + h] = acc[p];
  }
}

// ---- conv1 scatter + relu ---------------------------------------------------
__global__ void k_conv1_init(const float* __restrict__ x1,
                             const float* __restrict__ dis,
                             const float* __restrict__ b1,
                             float* __restrict__ out1) {
  int t = blockIdx.x * 256 + threadIdx.x;         // t = i*64 + h
  if (t < NN * DF) {
    int i = t >> 6, h = t & 63;
    float d = dis[i];
    out1[t] = b1[h] + d * d * x1[t];              // self-loop term + bias
  }
}

__global__ void k_edge_scatter(const int* __restrict__ ei,
                               const float* __restrict__ ew,
                               const float* __restrict__ dis,
                               const float* __restrict__ src,
                               float* __restrict__ dst) {
  int t = blockIdx.x * 256 + threadIdx.x;         // t = e*64 + h
  if (t < NE * DF) {
    int e = t >> 6, h = t & 63;
    int r = ei[e], c = ei[NE + e];
    float norm = dis[r] * ew[e] * dis[c];
    atomicAdd(&dst[(size_t)c * DF + h], norm * src[(size_t)r * DF + h]);
  }
}

__global__ void k_relu_ginit(const float* __restrict__ out1,
                             const float* __restrict__ dis,
                             float* __restrict__ hbuf,
                             float* __restrict__ g) {
  int t = blockIdx.x * 256 + threadIdx.x;
  if (t < NN * DF) {
    int i = t >> 6;
    float v = fmaxf(out1[t], 0.f);
    hbuf[t] = v;
    float d = dis[i];
    g[t] = d * d * v;                             // conv2 self-loop term
  }
}

// ---- pack to WMMA fragment layouts (ISA 7.12.2, f16 16x16x32) --------------
// A frag: lane row = l&15; vgpr v holds K = (v<4?0:16) + 8*(l>>4) + 2*(v&3), +1
__global__ void k_pack_A(const float* __restrict__ g, unsigned int* __restrict__ Ap) {
  int t = blockIdx.x * 256 + threadIdx.x;         // 512 mtiles * 2 ksteps * 32 * 8
  if (t < 512 * 2 * 32 * 8) {
    int v = t & 7, lane = (t >> 3) & 31, ks = (t >> 8) & 1, mt = t >> 9;
    int row = mt * 16 + (lane & 15);
    int k = ks * 32 + ((v < 4) ? 0 : 16) + 8 * (lane >> 4) + 2 * (v & 3);
    const float* gr = g + (size_t)row * DF + k;
    Ap[t] = pack2h(gr[0], gr[1]);
  }
}

// B frag: lane col = l&15; vgpr v holds K = (v<4?0:16) + 8*(l>>4) + 2*(v&3), +1
__global__ void k_pack_B(const float* __restrict__ W2, unsigned int* __restrict__ Bp) {
  int t = blockIdx.x * 256 + threadIdx.x;         // 2 ksteps * 1024 jtiles * 32 * 8
  if (t < 2 * 1024 * 32 * 8) {
    int v = t & 7, lane = (t >> 3) & 31, jt = (t >> 8) & 1023, ks = t >> 18;
    int k = ks * 32 + ((v < 4) ? 0 : 16) + 8 * (lane >> 4) + 2 * (v & 3);
    int n = jt * 16 + (lane & 15);
    Bp[t] = pack2h(W2[(size_t)k * NE + n], W2[(size_t)(k + 1) * NE + n]);
  }
}

// ---- main GEMM: out = g @ W2 + b2  ([8192,64]@[64,16384]) ------------------
// Each wave: one 16-row m-tile, sweeps 4 adjacent 16-col j-tiles reusing the
// A fragment in registers. K=64 = two v_wmma_f32_16x16x32_f16 per tile.
// Bias folded into accumulator init (all 8 D vgprs of a lane share column n).
__global__ void k_gemm(const unsigned int* __restrict__ Ap,
                       const unsigned int* __restrict__ Bp,
                       const float* __restrict__ b2,
                       float* __restrict__ out) {
  int wave = (blockIdx.x * blockDim.x + threadIdx.x) >> 5;
  int lane = threadIdx.x & 31;
  int jq = wave & 255;                            // 256 groups of 4 j-tiles
  int mt = wave >> 8;                             // 512 row tiles
  int jt0 = jq * 4;

  HU a0, a1;
  a0.u = *(const v8u*)(Ap + (size_t)((mt * 2 + 0) * 32 + lane) * 8);
  a1.u = *(const v8u*)(Ap + (size_t)((mt * 2 + 1) * 32 + lane) * 8);

  int m0 = mt * 16 + (lane >> 4) * 8;
  int nl = lane & 15;
  float* obase = out + (size_t)m0 * NE + nl;

#pragma unroll
  for (int jj = 0; jj < 4; ++jj) {
    int jt = jt0 + jj;
    HU b0, b1;
    b0.u = *(const v8u*)(Bp + (size_t)((0 * 1024 + jt) * 32 + lane) * 8);
    b1.u = *(const v8u*)(Bp + (size_t)((1 * 1024 + jt) * 32 + lane) * 8);

    float bias = b2[jt * 16 + nl];
    v8f c = { bias, bias, bias, bias, bias, bias, bias, bias };

    c = __builtin_amdgcn_wmma_f32_16x16x32_f16(false, a0.h, false, b0.h,
                                               (short)0, c, false, false);
    c = __builtin_amdgcn_wmma_f32_16x16x32_f16(false, a1.h, false, b1.h,
                                               (short)0, c, false, false);

    float* o = obase + jt * 16;
#pragma unroll
    for (int r = 0; r < 8; ++r) o[(size_t)r * NE] = c[r];
  }
}

// ---------------------------------------------------------------------------
extern "C" void kernel_launch(void* const* d_in, const int* in_sizes, int n_in,
                              void* d_out, int out_size, void* d_ws, size_t ws_size,
                              hipStream_t stream) {
  const int*   nodes = (const int*)  d_in[0];
  const int*   ei    = (const int*)  d_in[1];   // [2,E]: row = ei[e], col = ei[E+e]
  const float* ew    = (const float*)d_in[2];
  const int*   nbr   = (const int*)  d_in[3];   // [N,4]
  const float* emb   = (const float*)d_in[4];   // [N,64]
  const float* W1    = (const float*)d_in[5];   // [320,64]
  const float* b1    = (const float*)d_in[6];   // [64]
  const float* W2    = (const float*)d_in[7];   // [64,16384]
  const float* b2    = (const float*)d_in[8];   // [16384]
  float* out = (float*)d_out;                   // [8192,16384]

  char* ws = (char*)d_ws;
  float*        dis  = (float*)(ws + 0);                    // 32 KB (deg, then 1/sqrt)
  float*        x1   = (float*)(ws + (1u << 16));           // 2 MB
  float*        out1 = (float*)(ws + (1u << 16) + (1u << 21));
  float*        hbuf = (float*)(ws + (1u << 16) + 2u * (1u << 21));
  float*        g    = (float*)(ws + (1u << 16) + 3u * (1u << 21));
  unsigned int* Ap   = (unsigned int*)(ws + (1u << 16) + 4u * (1u << 21));  // 1 MB
  unsigned int* Bp   = (unsigned int*)(ws + (1u << 16) + 4u * (1u << 21) + (1u << 20)); // 2 MB

  k_deg_init   <<<NN / 256, 256, 0, stream>>>(dis);
  k_deg_scatter<<<NE / 256, 256, 0, stream>>>(ei, ew, dis);
  k_dis        <<<NN / 256, 256, 0, stream>>>(dis);

  k_x1<<<NN / 32, 256, 0, stream>>>(emb, nodes, nbr, W1, x1);

  k_conv1_init  <<<NN * DF / 256, 256, 0, stream>>>(x1, dis, b1, out1);
  k_edge_scatter<<<NE * DF / 256, 256, 0, stream>>>(ei, ew, dis, x1, out1);

  k_relu_ginit  <<<NN * DF / 256, 256, 0, stream>>>(out1, dis, hbuf, g);
  k_edge_scatter<<<NE * DF / 256, 256, 0, stream>>>(ei, ew, dis, hbuf, g);

  k_pack_A<<<512 * 2 * 32 * 8 / 256, 256, 0, stream>>>(g, Ap);
  k_pack_B<<<2 * 1024 * 32 * 8 / 256, 256, 0, stream>>>(W2, Bp);

  // 512 m-tiles * 256 j-quads = 131072 waves, 8 waves (256 threads) per block
  k_gemm<<<512 * 256 / 8, 256, 0, stream>>>(Ap, Bp, b2, out);
}
